// Attention_39006892982631
// MI455X (gfx1250) — matrix operand order repaired and checked by reading, not compiled, not run on hardware
//
#include <hip/hip_runtime.h>

typedef __attribute__((ext_vector_type(16))) __bf16 v16bf;
typedef __attribute__((ext_vector_type(8)))  __bf16 v8bf;
typedef __attribute__((ext_vector_type(2)))  __bf16 v2bf;
typedef __attribute__((ext_vector_type(8)))  float  v8f;
typedef __attribute__((ext_vector_type(4)))  unsigned int u32x4;
typedef __attribute__((ext_vector_type(8)))  int i32x8;
typedef __attribute__((ext_vector_type(4)))  int i32x4;

#define WMMA_BF16(a, b, c) \
  __builtin_amdgcn_wmma_f32_16x16x32_bf16(false, (a), false, (b), (short)0, (c), false, false)

constexpr int BB = 4;     // batch
constexpr int SS = 4096;  // sequence
constexpr int DD = 256;   // head dim

__device__ __forceinline__ float fast_exp2(float x) {
#if __has_builtin(__builtin_amdgcn_exp2f)
  return __builtin_amdgcn_exp2f(x);
#else
  return exp2f(x);
#endif
}

// A-fragment element e (0..15) -> K offset within a 32-wide K slab (16-bit A, 16x32), contiguous: k = 16*half + e
__device__ __forceinline__ int a_kof(int e, int half) {
  return half * 16 + e;
}

// ---------------------------------------------------------------------------
// Tensor Data Mover: stage a 2D tile (rows x rowElems of bf16) from global
// memory into LDS. D# built per CDNA5 ISA. Wave-uniform;
// EXEC is ignored by TDM. Completion tracked with TENSORcnt.
// ---------------------------------------------------------------------------
__device__ __forceinline__ void tdm_load_tile_bf16(
    unsigned lds_byte_off, const void* gaddr, unsigned rowElems, unsigned rows,
    unsigned tensorRows, unsigned rowStrideElems) {
  unsigned long long ga = (unsigned long long)gaddr;
  u32x4 g0;
  g0[0] = 1u;                                         // count=1 user descriptor
  g0[1] = lds_byte_off;                               // lds_addr (bytes)
  g0[2] = (unsigned)(ga & 0xffffffffu);               // global_addr[31:0]
  g0[3] = (unsigned)((ga >> 32) & 0x1ffffffu)         // global_addr[56:32]
          | (2u << 30);                               // type=2 ("image")
  i32x8 g1;
  g1[0] = (1 << 16);                                  // data_size=1 (2 bytes)
  g1[1] = (int)((rowElems & 0xffffu) << 16);          // tensor_dim0[15:0] @ bits63:48
  g1[2] = (int)(((rowElems >> 16) & 0xffffu)          // tensor_dim0[31:16]
          | ((tensorRows & 0xffffu) << 16));          // tensor_dim1[15:0]
  g1[3] = (int)(((tensorRows >> 16) & 0xffffu)        // tensor_dim1[31:16]
          | ((rowElems & 0xffffu) << 16));            // tile_dim0
  g1[4] = (int)(rows & 0xffffu);                      // tile_dim1 (tile_dim2=0)
  g1[5] = (int)rowStrideElems;                        // tensor_dim0_stride[31:0]
  g1[6] = 0;                                          // stride hi / dim1_stride lo
  g1[7] = 0;
  i32x4 z4 = {0, 0, 0, 0};
#if __clang_major__ >= 23
  i32x8 z8 = {0, 0, 0, 0, 0, 0, 0, 0};
  __builtin_amdgcn_tensor_load_to_lds(g0, g1, z4, z4, z8, 0);
#else
  __builtin_amdgcn_tensor_load_to_lds(g0, g1, z4, z4, 0);
#endif
}

// ---------------------------------------------------------------------------
// Projection: out[m, n] = (sum_d X[m, d] * W[n, d] + bias[n]) * outScale
// One wave computes a 16x16 output tile, K-loop of 8 bf16 WMMAs.
// ---------------------------------------------------------------------------
__global__ __launch_bounds__(256) void proj_kernel(
    const float* __restrict__ X, const float* __restrict__ W,
    const float* __restrict__ bias, __bf16* __restrict__ out, float outScale) {
  const int lane = threadIdx.x & 31;
  const int wave = threadIdx.x >> 5;
  const int half = lane >> 4;
  const int lr   = lane & 15;

  const long m_base = (long)(blockIdx.x * 8 + wave) * 16;
  const int  n_base = blockIdx.y * 16;

  const float* xrow = X + (m_base + lr) * DD;
  const float* wrow = W + (long)(n_base + lr) * DD;

  v8f acc = {};
  for (int k0 = 0; k0 < DD; k0 += 32) {
    v16bf a, b;
#pragma unroll
    for (int v = 0; v < 8; ++v) {
      const int e = 2 * v;
      const int k = a_kof(e, half);
      float2 f = *(const float2*)(xrow + k0 + k);
      a[e]     = (__bf16)f.x;
      a[e + 1] = (__bf16)f.y;
    }
    const float* wp = wrow + k0 + half * 16;
#pragma unroll
    for (int e = 0; e < 16; e += 4) {
      float4 f = *(const float4*)(wp + e);
      b[e] = (__bf16)f.x; b[e + 1] = (__bf16)f.y;
      b[e + 2] = (__bf16)f.z; b[e + 3] = (__bf16)f.w;
    }
    acc = WMMA_BF16(a, b, acc);
  }

  const float bv = bias[n_base + lr];
#pragma unroll
  for (int r = 0; r < 8; ++r) {
    const long row = m_base + r + 8 * half;
    out[row * DD + n_base + lr] = (__bf16)((acc[r] + bv) * outScale);
  }
}

// ---------------------------------------------------------------------------
// Flash attention over bf16 Q'/K'/V'. Scale AND log2(e) folded into Q', so the
// whole softmax runs in base-2 (v_exp_f32 is natively exp2). f32 output.
// Block: 128 threads = 4 waves, 64 queries (16 per wave). Key blocks of 32.
// K-block staged by TDM (tensor_load_to_lds); V-block transposed manually.
// ---------------------------------------------------------------------------
__global__ __launch_bounds__(128) void attn_kernel(
    const __bf16* __restrict__ Qp, const __bf16* __restrict__ Kp,
    const __bf16* __restrict__ Vp, float* __restrict__ Out) {
  __shared__ __bf16 ldsK[32 * 256];     // [key][dim], row-major (TDM dest)
  __shared__ __bf16 ldsVt[256 * 32];    // [dim][key], transposed
  __shared__ __bf16 ldsP[4][16 * 32];   // per-wave P bounce: [row][key]

  const int lane = threadIdx.x & 31;
  const int wave = threadIdx.x >> 5;
  const int half = lane >> 4;
  const int lr   = lane & 15;
  const int b    = blockIdx.y;
  const long baseRow = (long)b * SS;
  const int  q0 = blockIdx.x * 64 + wave * 16;

  // --- preload Q A-fragments: 8 slabs of 16x32 bf16 ---
  v16bf aq[8];
  const __bf16* qrow = Qp + (baseRow + q0 + lr) * DD;
#pragma unroll
  for (int s = 0; s < 8; ++s) {
#pragma unroll
    for (int v = 0; v < 8; ++v) {
      const int e = 2 * v;
      const int k = a_kof(e, half);
      v2bf t = *(const v2bf*)(qrow + s * 32 + k);
      aq[s][e] = t.x; aq[s][e + 1] = t.y;
    }
  }

  float mstate[8], lstate[8];
  v8f o[16];
#pragma unroll
  for (int r = 0; r < 8; ++r) { mstate[r] = -INFINITY; lstate[r] = 0.0f; }
#pragma unroll
  for (int t = 0; t < 16; ++t) o[t] = (v8f){};

  const unsigned ldsK_off = (unsigned)(uintptr_t)(&ldsK[0]);  // LDS byte offset

  for (int kb = 0; kb < SS; kb += 32) {
    __syncthreads();  // previous iteration's LDS readers are done

    // --- TDM: DMA the 32x256 bf16 K tile into LDS (wave 0 issues) ---
    if (wave == 0) {
      tdm_load_tile_bf16(ldsK_off, Kp + (baseRow + kb) * DD,
                         DD, 32, SS, DD);
      __builtin_amdgcn_s_wait_tensorcnt(0);
    }

    // --- V staged transposed by all threads (TDM cannot transpose) ---
    {
      const uint4* vsrc = (const uint4*)(Vp + (baseRow + kb) * DD);
      for (int i = threadIdx.x; i < 32 * 256 / 8; i += 128) {
        const int key = (i * 8) / 256;
        const int dim = (i * 8) % 256;
        uint4 f = vsrc[i];
        const __bf16* e8 = (const __bf16*)&f;
#pragma unroll
        for (int j = 0; j < 8; ++j) ldsVt[(dim + j) * 32 + key] = e8[j];
      }
    }

    // --- prefetch next key block while this one computes ---
    if (kb + 32 < SS) {
      __builtin_prefetch(Kp + (baseRow + kb + 32) * DD + lane * 16, 0, 1);
      __builtin_prefetch(Vp + (baseRow + kb + 32) * DD + lane * 16, 0, 1);
    }
    __syncthreads();

    // --- S = Q' K'^T for two 16-key subtiles ---
    v8f sc[2];
    sc[0] = (v8f){}; sc[1] = (v8f){};
#pragma unroll
    for (int nt = 0; nt < 2; ++nt) {
      const __bf16* krow = ldsK + (nt * 16 + lr) * DD;  // B: N = lr (key row)
#pragma unroll
      for (int s = 0; s < 8; ++s) {
        const v8bf* p = (const v8bf*)(krow + s * 32 + half * 16);
        v8bf x0 = p[0], x1 = p[1];
        v16bf bk;
#pragma unroll
        for (int j = 0; j < 8; ++j) { bk[j] = x0[j]; bk[8 + j] = x1[j]; }
        sc[nt] = WMMA_BF16(aq[s], bk, sc[nt]);
      }
    }

    // --- online softmax in base-2 (row = r + 8*half) ---
#pragma unroll
    for (int r = 0; r < 8; ++r) {
      float mx = fmaxf(sc[0][r], sc[1][r]);
#pragma unroll
      for (int off = 1; off < 16; off <<= 1) mx = fmaxf(mx, __shfl_xor(mx, off, 32));
      const float mnew  = fmaxf(mstate[r], mx);
      const float alpha = fast_exp2(mstate[r] - mnew);
      mstate[r] = mnew;
      const float p0 = fast_exp2(sc[0][r] - mnew);
      const float p1 = fast_exp2(sc[1][r] - mnew);
      sc[0][r] = p0; sc[1][r] = p1;
      float rs = p0 + p1;
#pragma unroll
      for (int off = 1; off < 16; off <<= 1) rs += __shfl_xor(rs, off, 32);
      lstate[r] = lstate[r] * alpha + rs;
#pragma unroll
      for (int t = 0; t < 16; ++t) o[t][r] *= alpha;
    }

    // --- P: C-layout -> A-layout via per-wave LDS bounce ---
    __bf16* pw = &ldsP[wave][0];
#pragma unroll
    for (int r = 0; r < 8; ++r) {
      const int row = r + 8 * half;
      pw[row * 32 + lr]      = (__bf16)sc[0][r];
      pw[row * 32 + 16 + lr] = (__bf16)sc[1][r];
    }
    asm volatile("s_wait_dscnt 0" ::: "memory");  // cross-lane LDS RAW within wave

    v16bf ap;
#pragma unroll
    for (int v = 0; v < 8; ++v) {
      const int e = 2 * v;
      const int k = a_kof(e, half);
      v2bf t = *(const v2bf*)(pw + lr * 32 + k);
      ap[e] = t.x; ap[e + 1] = t.y;
    }

    // --- O += P @ V over 16 output dim-tiles ---
#pragma unroll
    for (int t = 0; t < 16; ++t) {
      const v8bf* p = (const v8bf*)(ldsVt + (t * 16 + lr) * 32 + half * 16);
      v8bf x0 = p[0], x1 = p[1];
      v16bf bv;
#pragma unroll
      for (int j = 0; j < 8; ++j) { bv[j] = x0[j]; bv[8 + j] = x1[j]; }
      o[t] = WMMA_BF16(ap, bv, o[t]);
    }
  }

  // --- normalize and write out (f32) ---
#pragma unroll
  for (int r = 0; r < 8; ++r) {
    const float inv = 1.0f / lstate[r];
    float* orow = Out + (baseRow + q0 + r + 8 * half) * DD;
#pragma unroll
    for (int t = 0; t < 16; ++t) orow[t * 16 + lr] = o[t][r] * inv;
  }
}

extern "C" void kernel_launch(void* const* d_in, const int* in_sizes, int n_in,
                              void* d_out, int out_size, void* d_ws, size_t ws_size,
                              hipStream_t stream) {
  const float* q  = (const float*)d_in[0];
  const float* k  = (const float*)d_in[1];
  const float* v  = (const float*)d_in[2];
  const float* Wq = (const float*)d_in[3];
  const float* bq = (const float*)d_in[4];
  const float* Wk = (const float*)d_in[5];
  const float* bk = (const float*)d_in[6];
  const float* Wv = (const float*)d_in[7];
  const float* bv = (const float*)d_in[8];
  float* out = (float*)d_out;

  const size_t N = (size_t)BB * SS * DD;
  __bf16* qp = (__bf16*)d_ws;
  __bf16* kp = qp + N;
  __bf16* vp = kp + N;

  // 1/sqrt(256) * log2(e): softmax runs in base-2 downstream.
  const float qscale = 1.4426950408889634f / 16.0f;
  dim3 pgrid(BB * SS / (16 * 8), DD / 16);
  proj_kernel<<<pgrid, 256, 0, stream>>>(q, Wq, bq, qp, qscale);
  proj_kernel<<<pgrid, 256, 0, stream>>>(k, Wk, bk, kp, 1.0f);
  proj_kernel<<<pgrid, 256, 0, stream>>>(v, Wv, bv, vp, 1.0f);

  dim3 agrid(SS / 64, BB);
  attn_kernel<<<agrid, 128, 0, stream>>>(qp, kp, vp, out);
}
